// LSTMCell_38182259261440
// MI455X (gfx1250) — matrix operand order repaired and checked by reading, not compiled, not run on hardware
//
#include <hip/hip_runtime.h>
#include <hip/hip_bf16.h>

// ---------------- problem constants (from reference) ----------------
#define T_STEPS 4096
#define BATCH   8
#define DIN     512
#define NH      512
#define NG      2048          // 4*NH gate columns
#define KTOT    1024          // DIN + NH merged K
#define NBLK    32            // persistent workgroups; each owns 16 h-cols
#define NTHREADS 128          // 4 waves (wave32) -> 1 gate tile per wave
#define HCOLS   16            // h columns per workgroup
#define LDK     1032          // padded K stride (elements); 2064B row -> conflict-free b128

typedef __attribute__((ext_vector_type(16))) __bf16 bf16x16;
typedef __attribute__((ext_vector_type(8)))  __bf16 bf16x8;
typedef __attribute__((ext_vector_type(8)))  float  f32x8;

#define LOG2E  1.4426950408889634f
#define LOG2E2 2.8853900817779268f

// Branch-free sigmoid: 1/(1+2^(-x*log2e)); exp2 arg unbounded but
// v_exp_f32 saturates to +inf/0 which yields the correct 0/1 limits.
__device__ __forceinline__ float fast_sigmoid(float x) {
    float e = __builtin_amdgcn_exp2f(-LOG2E * x);
    return __builtin_amdgcn_rcpf(1.0f + e);
}
// Branch-free tanh: t = 2^(-2|x|*log2e) in (0,1]; tanh|x| = (1-t)/(1+t).
__device__ __forceinline__ float fast_tanh(float x) {
    float ax = __builtin_fabsf(x);
    float e  = __builtin_amdgcn_exp2f(-LOG2E2 * ax);
    float t  = (1.0f - e) * __builtin_amdgcn_rcpf(1.0f + e);
    return __builtin_copysignf(t, x);
}

// ---------------- one-time: pack [Wi;Wh] transposed to bf16, fold biases ----
extern "C" __global__ void lstm_pack_weights(const float* __restrict__ Wi,
                                             const float* __restrict__ Wh,
                                             const float* __restrict__ bi,
                                             const float* __restrict__ bh,
                                             __bf16* __restrict__ Wt,
                                             float* __restrict__ bias)
{
    size_t idx = (size_t)blockIdx.x * blockDim.x + threadIdx.x;
    if (idx < (size_t)NG * KTOT) {
        int n = (int)(idx >> 10);        // output column 0..2047
        int k = (int)(idx & (KTOT - 1)); // merged K 0..1023
        float v = (k < DIN) ? Wi[(size_t)k * NG + n]
                            : Wh[(size_t)(k - DIN) * NG + n];
        Wt[idx] = (__bf16)v;
    }
    if (idx < NG) bias[idx] = bi[idx] + bh[idx];
}

// ---------------- one-time per launch: zero step counters, seed h(bf16) -----
extern "C" __global__ void lstm_init_state(const float* __restrict__ h0,
                                           __bf16* __restrict__ hbf,
                                           int* __restrict__ cnt)
{
    int i = blockIdx.x * blockDim.x + threadIdx.x;
    if (i < BATCH * NH) hbf[i] = (__bf16)h0[i];   // buffer 0 = initial h
    if (i < T_STEPS)    cnt[i] = 0;
}

// ---------------- persistent cooperative recurrence -------------------------
extern "C" __global__ __launch_bounds__(NTHREADS, 1)
void lstm_recurrent(const float* __restrict__ x,
                    const float* __restrict__ h0,
                    const __bf16* __restrict__ Wt,
                    const float* __restrict__ bias,
                    __bf16* __restrict__ hbf,     // [2][BATCH][NH] double buffer
                    int* __restrict__ cnt,        // [T_STEPS] arrival counters
                    float* __restrict__ out)      // [T][B][NH] ++ [2][B][NH]
{
    extern __shared__ char smem[];
    __bf16* sB    = (__bf16*)smem;                               // 64 x LDK weights
    __bf16* sA    = (__bf16*)(smem + (size_t)64 * LDK * 2);      // 16 x LDK activations
    float*  gbuf  = (float*)(smem + (size_t)(64 + 16) * LDK * 2);// [4][16][16] gate pre
    float*  scell = gbuf + 4 * 16 * 16;                          // [8][16] cell slice

    const int w    = blockIdx.x;        // 0..31: owns h cols [w*16, w*16+16)
    const int tid  = threadIdx.x;
    const int lane = tid & 31;
    const int wave = tid >> 5;          // gate id 0..3 (f,i,o,g)
    const int hi   = lane >> 4;
    const int ml   = lane & 15;

    // Preload this workgroup's weight slice into LDS, 8 bf16 at a time.
    // Local col cl = gate*16 + j  ->  global col n = gate*512 + w*16 + j.
    for (int e8 = tid; e8 < (64 * KTOT) / 8; e8 += NTHREADS) {
        int cl = e8 >> 7;                 // local col
        int k  = (e8 & 127) * 8;          // K offset
        int n  = (cl >> 4) * NH + w * HCOLS + (cl & 15);
        *(bf16x8*)&sB[cl * LDK + k] =
            *(const bf16x8*)&Wt[(size_t)n * KTOT + k];
    }
    // Zero the A-matrix padding rows 8..15 once (batch is only 8).
    {
        const __bf16 z0 = (__bf16)0.0f;
        bf16x8 z = { z0, z0, z0, z0, z0, z0, z0, z0 };
        for (int e8 = tid; e8 < (8 * KTOT) / 8; e8 += NTHREADS) {
            int m = 8 + (e8 >> 7);
            int k = (e8 & 127) * 8;
            *(bf16x8*)&sA[m * LDK + k] = z;
        }
    }
    // Cell slice from hidden0[1].
    if (tid < BATCH * HCOLS) {
        int b = tid >> 4, c = tid & 15;
        scell[tid] = h0[1 * BATCH * NH + b * NH + w * HCOLS + c];
    }
    const float bv = bias[wave * NH + w * HCOLS + ml]; // per-lane output-col bias
    __syncthreads();

    for (int t = 0; t < T_STEPS; ++t) {
        // ---- stage A rows 0..7 = [ x[t] | h(t-1) ], vectorized b128 ----
        const float4*  xt4  = (const float4*)(x + (size_t)t * BATCH * DIN);
        const __bf16*  hrd  = hbf + (size_t)(t & 1) * BATCH * NH;
        for (int e8 = tid; e8 < (BATCH * DIN) / 8; e8 += NTHREADS) {
            float4 a = xt4[e8 * 2], b = xt4[e8 * 2 + 1];
            bf16x8 v = { (__bf16)a.x, (__bf16)a.y, (__bf16)a.z, (__bf16)a.w,
                         (__bf16)b.x, (__bf16)b.y, (__bf16)b.z, (__bf16)b.w };
            int m = e8 >> 6;              // 64 chunks of 8 per 512-row
            int k = (e8 & 63) * 8;
            *(bf16x8*)&sA[m * LDK + k] = v;
        }
        for (int e8 = tid; e8 < (BATCH * NH) / 8; e8 += NTHREADS) {
            int m = e8 >> 6;
            int k = (e8 & 63) * 8;
            *(bf16x8*)&sA[m * LDK + DIN + k] =
                *(const bf16x8*)&hrd[e8 * 8];
        }
        // Prefetch next timestep's x (recurrence-independent): 128 threads
        // x 128B cachelines = 16KB, overlapped with this step's WMMAs.
        if (t + 1 < T_STEPS)
            __builtin_prefetch(x + (size_t)(t + 1) * BATCH * DIN + tid * 32, 0, 1);
        __syncthreads();

        // ---- one 16x16 gate tile per wave, K=1024 chained WMMAs ----
        f32x8 acc = {0.f, 0.f, 0.f, 0.f, 0.f, 0.f, 0.f, 0.f};
        const __bf16* pa = sA + ml * LDK + hi * 8;                 // A frag base
        const __bf16* pb = sB + (wave * 16 + ml) * LDK + hi * 8;   // B frag base
        #pragma unroll 4
        for (int ks = 0; ks < KTOT; ks += 32) {
            bf16x8 a0 = *(const bf16x8*)(pa + ks);
            bf16x8 a1 = *(const bf16x8*)(pa + ks + 16);
            bf16x8 b0 = *(const bf16x8*)(pb + ks);
            bf16x8 b1 = *(const bf16x8*)(pb + ks + 16);
            bf16x16 a = __builtin_shufflevector(a0, a1,
                0, 1, 2, 3, 4, 5, 6, 7, 8, 9, 10, 11, 12, 13, 14, 15);
            bf16x16 b = __builtin_shufflevector(b0, b1,
                0, 1, 2, 3, 4, 5, 6, 7, 8, 9, 10, 11, 12, 13, 14, 15);
            acc = __builtin_amdgcn_wmma_f32_16x16x32_bf16(
                false, a, false, b, (short)0, acc, false, false);
        }
        // D layout: lane l, vgpr r -> row r + 8*(l>=16), col l&15.
        #pragma unroll
        for (int r = 0; r < 8; ++r)
            gbuf[(wave * 16 + (r + hi * 8)) * 16 + ml] = acc[r] + bv;
        __syncthreads();

        // ---- elementwise gate math: thread -> (batch b, local col c) ----
        {
            int b = tid >> 4, c = tid & 15;
            float fpre = gbuf[(0 * 16 + b) * 16 + c];
            float ipre = gbuf[(1 * 16 + b) * 16 + c];
            float opre = gbuf[(2 * 16 + b) * 16 + c];
            float gpre = gbuf[(3 * 16 + b) * 16 + c];
            float fg = fast_sigmoid(fpre);
            float ig = fast_sigmoid(ipre);
            float og = fast_sigmoid(opre);
            float gg = fast_tanh(gpre);
            float cell = fg * scell[tid] + ig * gg;
            scell[tid] = cell;
            float h = og * fast_tanh(cell);
            int gcol = w * HCOLS + c;
            out[(size_t)t * BATCH * NH + b * NH + gcol] = h;
            hbf[(size_t)((t + 1) & 1) * BATCH * NH + b * NH + gcol] = (__bf16)h;
            if (t == T_STEPS - 1) {
                size_t so = (size_t)T_STEPS * BATCH * NH;
                out[so + b * NH + gcol] = h;                  // final h
                out[so + BATCH * NH + b * NH + gcol] = cell;  // final cell
            }
        }
        __threadfence();   // release h writes device-wide
        __syncthreads();

        // ---- device-wide step barrier via per-step arrival counter ----
        if (tid == 0) {
            __hip_atomic_fetch_add(&cnt[t], 1, __ATOMIC_ACQ_REL,
                                   __HIP_MEMORY_SCOPE_AGENT);
            while (__hip_atomic_load(&cnt[t], __ATOMIC_ACQUIRE,
                                     __HIP_MEMORY_SCOPE_AGENT) < NBLK)
                __builtin_amdgcn_s_sleep(1);
        }
        __syncthreads();
    }
}

// ---------------- host entry -------------------------------------------------
extern "C" void kernel_launch(void* const* d_in, const int* in_sizes, int n_in,
                              void* d_out, int out_size, void* d_ws, size_t ws_size,
                              hipStream_t stream)
{
    const float* x  = (const float*)d_in[0];
    const float* h0 = (const float*)d_in[1];
    const float* Wi = (const float*)d_in[2];
    const float* bi = (const float*)d_in[3];
    const float* Wh = (const float*)d_in[4];
    const float* bh = (const float*)d_in[5];
    float* out = (float*)d_out;

    // workspace layout
    char*   ws   = (char*)d_ws;
    __bf16* Wt   = (__bf16*)ws;                                   // 4 MiB
    size_t  off  = (size_t)NG * KTOT * 2;
    float*  bias = (float*)(ws + off);    off += (size_t)NG * 4;  // 8 KiB
    __bf16* hbf  = (__bf16*)(ws + off);   off += (size_t)2 * BATCH * NH * 2; // 16 KiB
    int*    cnt  = (int*)(ws + off);                               // 16 KiB

    lstm_pack_weights<<<(NG * KTOT + 255) / 256, 256, 0, stream>>>(
        Wi, Wh, bi, bh, Wt, bias);
    lstm_init_state<<<(T_STEPS + 255) / 256, 256, 0, stream>>>(h0, hbf, cnt);

    size_t lds = (size_t)(64 + 16) * LDK * 2  // weight slice + A staging (bf16)
               + (size_t)4 * 16 * 16 * 4      // gate pre-activation buffer
               + (size_t)128 * 4;             // cell slice
    (void)hipFuncSetAttribute((const void*)lstm_recurrent,
                              hipFuncAttributeMaxDynamicSharedMemorySize,
                              (int)lds);
    lstm_recurrent<<<NBLK, NTHREADS, lds, stream>>>(
        x, h0, Wt, bias, hbf, cnt, out);
}
